// FastAttention_37443524886905
// MI455X (gfx1250) — matrix operand order repaired and checked by reading, not compiled, not run on hardware
//
#include <hip/hip_runtime.h>
#include <hip/hip_bf16.h>

typedef __attribute__((ext_vector_type(16))) __bf16 v16bf;
typedef __attribute__((ext_vector_type(8)))  __bf16 v8bf;
typedef __attribute__((ext_vector_type(8)))  float  v8f;

union V16U { v16bf v; v8bf h[2]; };

#define B_SZ 8
#define SEQ 8192
#define DIM 128
#define NORMALIZER 0.29730177875068026f   /* 128^-0.25 */
#define INV_SQRT_M 0.08838834764831845f   /* 1/sqrt(128) */
#define EPS_V 1e-6f

__device__ __forceinline__ v8f vzero8() {
  v8f z = {0.f,0.f,0.f,0.f,0.f,0.f,0.f,0.f};
  return z;
}

__device__ __forceinline__ v8f wmma_bf16(v16bf a, v16bf b, v8f c) {
  return __builtin_amdgcn_wmma_f32_16x16x32_bf16(
      /*neg_a=*/false, a, /*neg_b=*/false, b,
      /*c_mod=*/(short)0, c, /*reuse_a=*/false, /*reuse_b=*/false);
}

// ---------------------------------------------------------------------------
// Workspace zeroing
// ---------------------------------------------------------------------------
__global__ void favor_zero_kernel(float* __restrict__ p, int n) {
  int i = blockIdx.x * blockDim.x + threadIdx.x;
  if (i < n) p[i] = 0.f;
}

// ---------------------------------------------------------------------------
// Phase 1: kv[b] = ks'^T @ vs, ksum[b] = sum_k ks'  (masked by valid_lens)
// grid = 256 WGs (32 per batch), 256 threads (8 waves), each WG does 2x128 keys
// ---------------------------------------------------------------------------
__global__ __launch_bounds__(256) void favor_kv_kernel(
    const float* __restrict__ ks, const float* __restrict__ vs,
    const float* __restrict__ proj, const long long* __restrict__ valid_lens,
    float* __restrict__ kv_ws, float* __restrict__ ksum_ws) {
  extern __shared__ char smem[];
  __bf16* s_proj = (__bf16*)smem;               // [j][d] 128x128 bf16 (32KB)
  __bf16* s_vT   = (__bf16*)(smem + 32768);     // [v][key] 128x128 bf16 (32KB)
  __bf16* s_x    = (__bf16*)(smem + 65536);     // per-wave [16][128]; reused as ksp^T [m][key]
  __bf16* s_kspT = s_x;

  const int tid  = threadIdx.x;
  const int wave = tid >> 5;
  const int lane = tid & 31;
  const int half = lane >> 4;
  const int l16  = lane & 15;

  const int wgs_per_b = gridDim.x / B_SZ;
  const int b       = blockIdx.x / wgs_per_b;
  const int wg_in_b = blockIdx.x % wgs_per_b;
  const int iters   = SEQ / (wgs_per_b * 128);
  const long long vlen = valid_lens[b];

  // stage proj -> LDS bf16 (row-major [j][d]; B-operand reads are contiguous)
  for (int i = tid * 4; i < DIM * DIM; i += 1024) {
    float4 p = *(const float4*)(proj + i);
    s_proj[i]   = (__bf16)p.x; s_proj[i+1] = (__bf16)p.y;
    s_proj[i+2] = (__bf16)p.z; s_proj[i+3] = (__bf16)p.w;
  }

  v8f accKV[8];
  for (int t = 0; t < 8; ++t) accKV[t] = vzero8();
  v8f accS = vzero8();

  // ones-in-column-0 B operand (for ksum reduction)
  v16bf Bones;
  for (int i = 0; i < 16; ++i) Bones[i] = (l16 == 0) ? (__bf16)1.0f : (__bf16)0.0f;

  for (int it = 0; it < iters; ++it) {
    const int key0 = (wg_in_b * iters + it) * 128;

    // each wave stages its 16 scaled keys (bf16) + per-row |x|^2
    float hrow;
    {
      const float* src = ks + (((long long)b * SEQ) + key0 + wave * 16 + l16) * DIM + half * 64;
      __bf16* dst = s_x + wave * 2048 + l16 * 128 + half * 64;
      float ss = 0.f;
      for (int c = 0; c < 64; c += 4) {
        float4 x = *(const float4*)(src + c);
        float x0 = x.x * NORMALIZER, x1 = x.y * NORMALIZER;
        float x2 = x.z * NORMALIZER, x3 = x.w * NORMALIZER;
        ss += x0*x0 + x1*x1 + x2*x2 + x3*x3;
        dst[c]   = (__bf16)x0; dst[c+1] = (__bf16)x1;
        dst[c+2] = (__bf16)x2; dst[c+3] = (__bf16)x3;
      }
      hrow = 0.5f * (ss + __shfl_xor(ss, 16, 32));
    }
    // stage V chunk transposed: s_vT[v][key]
    for (int i = tid * 4; i < DIM * 128; i += 1024) {
      int key = i >> 7, v0 = i & 127;
      float4 x = *(const float4*)(vs + (((long long)b * SEQ) + key0 + key) * DIM + v0);
      s_vT[(v0    ) * 128 + key] = (__bf16)x.x;
      s_vT[(v0 + 1) * 128 + key] = (__bf16)x.y;
      s_vT[(v0 + 2) * 128 + key] = (__bf16)x.z;
      s_vT[(v0 + 3) * 128 + key] = (__bf16)x.w;
    }
    __syncthreads();

    // A operands of phi projection (16x32 bf16, lane=M; contiguous K order used here: K 0-15 vs 16-31)
    v16bf Areg[4];
    {
      const __bf16* xw = s_x + wave * 2048;
      for (int kc = 0; kc < 4; ++kc) {
        V16U a;
        const __bf16* ap = xw + l16 * 128 + kc * 32 + half * 16;
        a.h[0] = *(const v8bf*)(ap);
        a.h[1] = *(const v8bf*)(ap + 8);
        Areg[kc] = a.v;
      }
    }
    float hb[8];
    for (int r = 0; r < 8; ++r) hb[r] = __shfl(hrow, half * 8 + r, 32);
    __syncthreads();  // s_x consumed into registers; region reused as ksp^T

    // x_proj = x @ proj^T ; phi + mask ; store transposed [m][key]
    for (int t = 0; t < 8; ++t) {
      v8f acc = vzero8();
      for (int kc = 0; kc < 4; ++kc) {
        v16bf Bm = *(const v16bf*)(s_proj + (t * 16 + l16) * 128 + kc * 32 + half * 16);
        acc = wmma_bf16(Areg[kc], Bm, acc);
      }
      for (int r = 0; r < 8; ++r) {
        int row = r + half * 8;
        float val = __expf(acc[r] - hb[r]) * INV_SQRT_M;
        val = (((long long)(key0 + wave * 16 + row)) < vlen) ? val : 0.f;
        s_kspT[(t * 16 + l16) * 128 + wave * 16 + row] = (__bf16)val;
      }
    }
    __syncthreads();

    // kv (wave's 16 m-rows) += ksp^T @ V ; ksum via ones column
    {
      const __bf16* aw = s_kspT + (wave * 16) * 128;
      for (int kc = 0; kc < 4; ++kc) {
        V16U a;
        const __bf16* ap = aw + l16 * 128 + kc * 32 + half * 16;
        a.h[0] = *(const v8bf*)(ap);
        a.h[1] = *(const v8bf*)(ap + 8);
        for (int t = 0; t < 8; ++t) {
          v16bf Bm = *(const v16bf*)(s_vT + (t * 16 + l16) * 128 + kc * 32 + half * 16);
          accKV[t] = wmma_bf16(a.v, Bm, accKV[t]);
        }
        accS = wmma_bf16(a.v, Bones, accS);
      }
    }
    __syncthreads();
  }

  // merge partials into workspace
  float* kvb = kv_ws + b * DIM * DIM;
  for (int t = 0; t < 8; ++t)
    for (int r = 0; r < 8; ++r) {
      int m = wave * 16 + r + half * 8;
      atomicAdd(kvb + m * DIM + t * 16 + l16, accKV[t][r]);
    }
  if (l16 == 0)
    for (int r = 0; r < 8; ++r)
      atomicAdd(ksum_ws + b * DIM + wave * 16 + r + half * 8, accS[r]);
}

// ---------------------------------------------------------------------------
// Phase 2: out = (q' @ kv) / max(q' @ ksum, EPS)
// grid = 512 WGs (64 per batch), 256 threads, 128 queries per WG
// ---------------------------------------------------------------------------
__global__ __launch_bounds__(256) void favor_q_kernel(
    const float* __restrict__ qs, const float* __restrict__ proj,
    const float* __restrict__ kv_ws, const float* __restrict__ ksum_ws,
    float* __restrict__ out) {
  extern __shared__ char smem[];
  __bf16* s_proj = (__bf16*)smem;               // [j][d]
  __bf16* s_kvT  = (__bf16*)(smem + 32768);     // [v][m]
  __bf16* s_x    = (__bf16*)(smem + 65536);     // per-wave [16][128]; reused as q'
  float*  s_ksum = (float*)(smem + 98304);      // [128]

  const int tid  = threadIdx.x;
  const int wave = tid >> 5;
  const int lane = tid & 31;
  const int half = lane >> 4;
  const int l16  = lane & 15;

  const int blocks_per_b = gridDim.x / B_SZ;
  const int b  = blockIdx.x / blocks_per_b;
  const int q0 = (blockIdx.x % blocks_per_b) * 128;

  for (int i = tid * 4; i < DIM * DIM; i += 1024) {
    float4 p = *(const float4*)(proj + i);
    s_proj[i]   = (__bf16)p.x; s_proj[i+1] = (__bf16)p.y;
    s_proj[i+2] = (__bf16)p.z; s_proj[i+3] = (__bf16)p.w;
  }
  for (int i = tid * 4; i < DIM * DIM; i += 1024) {
    int m = i >> 7, v0 = i & 127;
    float4 x = *(const float4*)(kv_ws + b * DIM * DIM + i);
    s_kvT[(v0    ) * 128 + m] = (__bf16)x.x;
    s_kvT[(v0 + 1) * 128 + m] = (__bf16)x.y;
    s_kvT[(v0 + 2) * 128 + m] = (__bf16)x.z;
    s_kvT[(v0 + 3) * 128 + m] = (__bf16)x.w;
  }
  if (tid < DIM) s_ksum[tid] = ksum_ws[b * DIM + tid];

  float hrow;
  {
    const float* src = qs + (((long long)b * SEQ) + q0 + wave * 16 + l16) * DIM + half * 64;
    __bf16* dst = s_x + wave * 2048 + l16 * 128 + half * 64;
    float ss = 0.f;
    for (int c = 0; c < 64; c += 4) {
      float4 x = *(const float4*)(src + c);
      float x0 = x.x * NORMALIZER, x1 = x.y * NORMALIZER;
      float x2 = x.z * NORMALIZER, x3 = x.w * NORMALIZER;
      ss += x0*x0 + x1*x1 + x2*x2 + x3*x3;
      dst[c]   = (__bf16)x0; dst[c+1] = (__bf16)x1;
      dst[c+2] = (__bf16)x2; dst[c+3] = (__bf16)x3;
    }
    hrow = 0.5f * (ss + __shfl_xor(ss, 16, 32));
  }
  __syncthreads();

  v16bf Areg[4];
  {
    const __bf16* xw = s_x + wave * 2048;
    for (int kc = 0; kc < 4; ++kc) {
      V16U a;
      const __bf16* ap = xw + l16 * 128 + kc * 32 + half * 16;
      a.h[0] = *(const v8bf*)(ap);
      a.h[1] = *(const v8bf*)(ap + 8);
      Areg[kc] = a.v;
    }
  }
  float hb[8];
  for (int r = 0; r < 8; ++r) hb[r] = __shfl(hrow, half * 8 + r, 32);

  // q' tiles -> per-wave LDS region (row-major [q][m]), reusing s_x
  __bf16* qp = s_x + wave * 2048;
  for (int t = 0; t < 8; ++t) {
    v8f acc = vzero8();
    for (int kc = 0; kc < 4; ++kc) {
      v16bf Bm = *(const v16bf*)(s_proj + (t * 16 + l16) * 128 + kc * 32 + half * 16);
      acc = wmma_bf16(Areg[kc], Bm, acc);
    }
    for (int r = 0; r < 8; ++r) {
      int row = r + half * 8;
      qp[row * 128 + t * 16 + l16] = (__bf16)(__expf(acc[r] - hb[r]) * INV_SQRT_M);
    }
  }
  __syncthreads();

  // num = q' @ kv ; den = q' @ ksum (column-0 B operand)
  v8f accN[8];
  for (int t = 0; t < 8; ++t) accN[t] = vzero8();
  v8f accD = vzero8();
  for (int kc = 0; kc < 4; ++kc) {
    V16U a;
    const __bf16* ap = qp + l16 * 128 + kc * 32 + half * 16;
    a.h[0] = *(const v8bf*)(ap);
    a.h[1] = *(const v8bf*)(ap + 8);
    for (int t = 0; t < 8; ++t) {
      v16bf Bm = *(const v16bf*)(s_kvT + (t * 16 + l16) * 128 + kc * 32 + half * 16);
      accN[t] = wmma_bf16(a.v, Bm, accN[t]);
    }
    v16bf Bk;
    for (int i = 0; i < 16; ++i)
      Bk[i] = (l16 == 0) ? (__bf16)s_ksum[kc * 32 + half * 16 + i] : (__bf16)0.0f;
    accD = wmma_bf16(a.v, Bk, accD);
  }

  // epilogue: broadcast den (held in lanes 0/16, column 0), clamp, divide, store
  float* ob = out + (((long long)b * SEQ) + q0 + wave * 16) * DIM;
  for (int r = 0; r < 8; ++r) {
    float den = __shfl(accD[r], half * 16, 32);
    den = den < EPS_V ? EPS_V : den;
    float rden = 1.0f / den;
    int row = r + half * 8;
    for (int t = 0; t < 8; ++t)
      ob[row * DIM + t * 16 + l16] = accN[t][r] * rden;
  }
}

// ---------------------------------------------------------------------------
extern "C" void kernel_launch(void* const* d_in, const int* in_sizes, int n_in,
                              void* d_out, int out_size, void* d_ws, size_t ws_size,
                              hipStream_t stream) {
  const float*     qs    = (const float*)d_in[0];
  const float*     ks    = (const float*)d_in[1];
  const float*     vs    = (const float*)d_in[2];
  const float*     proj  = (const float*)d_in[3];
  const long long* vlens = (const long long*)d_in[4];  // int64 valid_lens
  float* out = (float*)d_out;

  float* kv_ws   = (float*)d_ws;                  // [B][128][128]
  float* ksum_ws = kv_ws + B_SZ * DIM * DIM;      // [B][128]

  const int nz = B_SZ * DIM * DIM + B_SZ * DIM;
  favor_zero_kernel<<<(nz + 255) / 256, 256, 0, stream>>>(kv_ws, nz);

  // Phase 1: 32 WGs per batch, each reduces 256 keys (2 x 128-key chunks)
  favor_kv_kernel<<<256, 256, 96 * 1024, stream>>>(ks, vs, proj, vlens, kv_ws, ksum_ws);

  // Phase 2: 64 WGs per batch, 128 queries each
  favor_q_kernel<<<512, 256, 96 * 1024 + 512, stream>>>(qs, proj, kv_ws, ksum_ws, out);
}
